// NomicExperts_42829413875911
// MI455X (gfx1250) — compile-verified, matrix operand order checked by reading, not run on hardware
//
#include <hip/hip_runtime.h>
#include <math.h>
#include <stdint.h>

// ---------------- problem constants ----------------
#define T_TOK 8192
#define HDIM  768
#define FDIM  3072
#define NEXP  8
#define TM    128
#define TN    128
#define TKS   32
#define LDSPAD 8
#define MAX_TILES (T_TOK * 2 / TM + NEXP)   // 136: worst-case tile-aligned segments

typedef __attribute__((ext_vector_type(16))) __bf16       v16bf;
typedef __attribute__((ext_vector_type(8)))  float        v8f;
typedef __attribute__((ext_vector_type(4)))  unsigned int u32x4;

union FragU { v16bf bf; u32x4 u4[2]; };

__device__ __forceinline__ unsigned short f2bf(float f) {
  unsigned int u = __float_as_uint(f);
  u += 0x7FFFu + ((u >> 16) & 1u);           // round-to-nearest-even
  return (unsigned short)(u >> 16);
}

__device__ __forceinline__ float gelu_exact(float v) {
  return 0.5f * v * (1.0f + erff(v * 0.70710678118654752440f));
}

// Async DMA: global -> LDS, 32 bytes per lane per matrix (two b128, ISA adds
// IOFFSET to both the global and the LDS address). Tracked by ASYNCcnt.
__device__ __forceinline__ void async_stage32(unsigned ldsOff, const void* g) {
  asm volatile(
      "global_load_async_to_lds_b128 %0, %1, off\n\t"
      "global_load_async_to_lds_b128 %0, %1, off offset:16"
      :: "v"(ldsOff), "v"((unsigned long long)(uintptr_t)g)
      : "memory");
}

__device__ __forceinline__ void async_wait0() {
  asm volatile("s_wait_asynccnt 0x0" ::: "memory");
}

__device__ __forceinline__ unsigned lds_off(const void* p) {
  // Generic LDS address: high dword = shared aperture, low dword = LDS byte addr.
  return (unsigned)(uintptr_t)p;
}

// ---------------- routing ----------------
__global__ void k_zero(int* cnt, int* nTiles) {
  int i = threadIdx.x;
  if (i < NEXP) cnt[i] = 0;
  if (i == 0) nTiles[0] = 0;
}

__global__ void k_route(const int* __restrict__ top_experts,
                        const float* __restrict__ top_weights,
                        int* __restrict__ cnt,
                        int* __restrict__ rowTok,
                        float* __restrict__ rowScale) {
  int t = blockIdx.x * blockDim.x + threadIdx.x;
  if (t >= T_TOK) return;
  int   e0 = top_experts[2 * t],  e1 = top_experts[2 * t + 1];
  float s0 = top_weights[2 * t],  s1 = top_weights[2 * t + 1];
  if (e0 == e1) {  // merge duplicate routing -> <=1 entry per (token, expert)
    int s = atomicAdd(&cnt[e0], 1);
    rowTok[e0 * T_TOK + s] = t; rowScale[e0 * T_TOK + s] = s0 + s1;
  } else {
    int a = atomicAdd(&cnt[e0], 1);
    rowTok[e0 * T_TOK + a] = t; rowScale[e0 * T_TOK + a] = s0;
    int b = atomicAdd(&cnt[e1], 1);
    rowTok[e1 * T_TOK + b] = t; rowScale[e1 * T_TOK + b] = s1;
  }
}

__global__ void k_tiles(const int* __restrict__ cnt,
                        int* __restrict__ tileExpert,
                        int* __restrict__ tileBase,
                        int* __restrict__ nTiles) {
  if (threadIdx.x == 0 && blockIdx.x == 0) {
    int nt = 0;
    for (int e = 0; e < NEXP; ++e)
      for (int b = 0; b < cnt[e]; b += TM) {
        tileExpert[nt] = e; tileBase[nt] = b; ++nt;
      }
    nTiles[0] = nt;  // <= MAX_TILES by construction
  }
}

// ---------------- precision prep ----------------
__global__ void k_prep_x_out(const float* __restrict__ x,
                             const float* __restrict__ bias,
                             unsigned short* __restrict__ xb,
                             float* __restrict__ out) {
  size_t i = (size_t)blockIdx.x * blockDim.x + threadIdx.x;
  if (i >= (size_t)T_TOK * HDIM) return;
  xb[i]  = f2bf(x[i]);
  out[i] = bias[i % HDIM];           // out pre-seeded with bias; GEMM2 atomically adds
}

__global__ void k_prep_w(const float* __restrict__ w1,
                         const float* __restrict__ w2,
                         unsigned short* __restrict__ w1b,
                         unsigned short* __restrict__ w2tb) {
  size_t i = (size_t)blockIdx.x * blockDim.x + threadIdx.x;
  size_t per = (size_t)FDIM * HDIM;
  if (i >= (size_t)NEXP * per) return;
  w1b[i] = f2bf(w1[i]);                           // keep [e][f][h]  (= B^T rows, contiguous K)
  size_t e = i / per, rem = i % per;
  size_t hcol = rem / FDIM, f = rem % FDIM;
  w2tb[i] = f2bf(w2[e * per + f * HDIM + hcol]);  // transpose -> [e][h][f] (contiguous K)
}

// ---------------- GEMM1: h = gelu(x_gather @ w1[e]^T), bf16 out ----------------
__global__ void __launch_bounds__(256)
k_gemm1(const unsigned short* __restrict__ xb,
        const unsigned short* __restrict__ w1b,
        unsigned short* __restrict__ hbuf,
        const int* __restrict__ rowTok,
        const int* __restrict__ cnt,
        const int* __restrict__ tileExpert,
        const int* __restrict__ tileBase,
        const int* __restrict__ nTiles) {
  const int mtile = blockIdx.x;
  if (mtile >= nTiles[0]) return;
  const int ntile = blockIdx.y;
  const int e     = tileExpert[mtile];
  const int base  = tileBase[mtile];
  const int valid = cnt[e] - base;

  __shared__ unsigned short As[2][TM][TKS + LDSPAD];
  __shared__ unsigned short Bs[2][TN][TKS + LDSPAD];
  __shared__ int tokS[TM];

  const int tid   = threadIdx.x;
  const int lane  = tid & 31;
  const int wave  = tid >> 5;
  const int waveM = wave & 3;    // 4 waves along M (32 rows each)
  const int waveN = wave >> 2;   // 2 waves along N (64 cols each)

  if (tid < TM) tokS[tid] = (tid < valid) ? rowTok[e * T_TOK + base + tid] : 0;
  __syncthreads();

  v8f acc[2][4];
  #pragma unroll
  for (int i = 0; i < 2; ++i)
    #pragma unroll
    for (int j = 0; j < 4; ++j)
      #pragma unroll
      for (int q = 0; q < 8; ++q) acc[i][j][q] = 0.0f;

  const int srow  = tid >> 1;
  const int shalf = tid & 1;
  const size_t aBase = (size_t)tokS[srow] * HDIM + (size_t)shalf * 16;
  const size_t bBase = ((size_t)e * FDIM + (size_t)ntile * TN + srow) * HDIM + (size_t)shalf * 16;
  const unsigned aOff[2] = { lds_off(&As[0][srow][shalf * 16]), lds_off(&As[1][srow][shalf * 16]) };
  const unsigned bOff[2] = { lds_off(&Bs[0][srow][shalf * 16]), lds_off(&Bs[1][srow][shalf * 16]) };

  const int koff = (lane >> 4) * 8;  // ISA 16-bit A/B layout: lane>=16 holds K 8..15 / 24..31
  const int rsel = lane & 15;

  auto stage = [&](int buf, int k0) {
    const unsigned short* ag = xb  + aBase + k0;
    const unsigned short* bg = w1b + bBase + k0;
    async_stage32(aOff[buf], ag);
    async_stage32(bOff[buf], bg);
    __builtin_prefetch(ag + TKS, 0, 1);   // hide HBM->L2 latency one more step ahead
    __builtin_prefetch(bg + TKS, 0, 1);
  };

  stage(0, 0);
  async_wait0();
  __syncthreads();

  int cur = 0;
  for (int k0 = 0; k0 < HDIM; k0 += TKS) {
    const int nxt = cur ^ 1;
    if (k0 + TKS < HDIM) stage(nxt, k0 + TKS);  // DMA overlaps WMMA below

    FragU af[2], bfm[4];
    #pragma unroll
    for (int mi = 0; mi < 2; ++mi) {
      const int r = waveM * 32 + mi * 16 + rsel;
      af[mi].u4[0] = *(const u32x4*)&As[cur][r][koff];
      af[mi].u4[1] = *(const u32x4*)&As[cur][r][koff + 16];
    }
    #pragma unroll
    for (int ni = 0; ni < 4; ++ni) {
      const int c = waveN * 64 + ni * 16 + rsel;
      bfm[ni].u4[0] = *(const u32x4*)&Bs[cur][c][koff];
      bfm[ni].u4[1] = *(const u32x4*)&Bs[cur][c][koff + 16];
    }
    #pragma unroll
    for (int mi = 0; mi < 2; ++mi)
      #pragma unroll
      for (int ni = 0; ni < 4; ++ni)
        acc[mi][ni] = __builtin_amdgcn_wmma_f32_16x16x32_bf16(
            false, af[mi].bf, false, bfm[ni].bf, (short)0, acc[mi][ni], false, false);

    async_wait0();
    __syncthreads();
    cur = nxt;
  }

  const size_t hrow0 = (size_t)mtile * TM;
  const int mhalf = (lane >> 4) * 8;   // C layout: lanes>=16 hold rows M=8..15 of subtile
  #pragma unroll
  for (int mi = 0; mi < 2; ++mi)
    #pragma unroll
    for (int ni = 0; ni < 4; ++ni) {
      const int n = ntile * TN + waveN * 64 + ni * 16 + rsel;
      #pragma unroll
      for (int r = 0; r < 8; ++r) {
        const int m = waveM * 32 + mi * 16 + mhalf + r;
        hbuf[(hrow0 + m) * FDIM + n] = f2bf(gelu_exact(acc[mi][ni][r]));
      }
    }
}

// ---------------- GEMM2: out[tok] += scale * (h @ w2[e]) ----------------
__global__ void __launch_bounds__(256)
k_gemm2(const unsigned short* __restrict__ hbuf,
        const unsigned short* __restrict__ w2tb,
        float* __restrict__ out,
        const int* __restrict__ rowTok,
        const float* __restrict__ rowScale,
        const int* __restrict__ cnt,
        const int* __restrict__ tileExpert,
        const int* __restrict__ tileBase,
        const int* __restrict__ nTiles) {
  const int mtile = blockIdx.x;
  if (mtile >= nTiles[0]) return;
  const int ntile = blockIdx.y;
  const int e     = tileExpert[mtile];
  const int base  = tileBase[mtile];
  const int valid = cnt[e] - base;

  __shared__ unsigned short As[2][TM][TKS + LDSPAD];
  __shared__ unsigned short Bs[2][TN][TKS + LDSPAD];
  __shared__ int   tokS[TM];
  __shared__ float sclS[TM];

  const int tid   = threadIdx.x;
  const int lane  = tid & 31;
  const int wave  = tid >> 5;
  const int waveM = wave & 3;
  const int waveN = wave >> 2;

  if (tid < TM) {
    bool ok = tid < valid;
    tokS[tid] = ok ? rowTok[e * T_TOK + base + tid] : 0;
    sclS[tid] = ok ? rowScale[e * T_TOK + base + tid] : 0.0f;
  }
  __syncthreads();

  v8f acc[2][4];
  #pragma unroll
  for (int i = 0; i < 2; ++i)
    #pragma unroll
    for (int j = 0; j < 4; ++j)
      #pragma unroll
      for (int q = 0; q < 8; ++q) acc[i][j][q] = 0.0f;

  const int srow  = tid >> 1;
  const int shalf = tid & 1;
  const size_t aBase = ((size_t)mtile * TM + srow) * FDIM + (size_t)shalf * 16;
  const size_t bBase = ((size_t)e * HDIM + (size_t)ntile * TN + srow) * FDIM + (size_t)shalf * 16;
  const unsigned aOff[2] = { lds_off(&As[0][srow][shalf * 16]), lds_off(&As[1][srow][shalf * 16]) };
  const unsigned bOff[2] = { lds_off(&Bs[0][srow][shalf * 16]), lds_off(&Bs[1][srow][shalf * 16]) };

  const int koff = (lane >> 4) * 8;
  const int rsel = lane & 15;

  auto stage = [&](int buf, int k0) {
    const unsigned short* ag = hbuf + aBase + k0;
    const unsigned short* bg = w2tb + bBase + k0;
    async_stage32(aOff[buf], ag);
    async_stage32(bOff[buf], bg);
    __builtin_prefetch(ag + TKS, 0, 1);
    __builtin_prefetch(bg + TKS, 0, 1);
  };

  stage(0, 0);
  async_wait0();
  __syncthreads();

  int cur = 0;
  for (int k0 = 0; k0 < FDIM; k0 += TKS) {
    const int nxt = cur ^ 1;
    if (k0 + TKS < FDIM) stage(nxt, k0 + TKS);

    FragU af[2], bfm[4];
    #pragma unroll
    for (int mi = 0; mi < 2; ++mi) {
      const int r = waveM * 32 + mi * 16 + rsel;
      af[mi].u4[0] = *(const u32x4*)&As[cur][r][koff];
      af[mi].u4[1] = *(const u32x4*)&As[cur][r][koff + 16];
    }
    #pragma unroll
    for (int ni = 0; ni < 4; ++ni) {
      const int c = waveN * 64 + ni * 16 + rsel;
      bfm[ni].u4[0] = *(const u32x4*)&Bs[cur][c][koff];
      bfm[ni].u4[1] = *(const u32x4*)&Bs[cur][c][koff + 16];
    }
    #pragma unroll
    for (int mi = 0; mi < 2; ++mi)
      #pragma unroll
      for (int ni = 0; ni < 4; ++ni)
        acc[mi][ni] = __builtin_amdgcn_wmma_f32_16x16x32_bf16(
            false, af[mi].bf, false, bfm[ni].bf, (short)0, acc[mi][ni], false, false);

    async_wait0();
    __syncthreads();
    cur = nxt;
  }

  const int mhalf = (lane >> 4) * 8;
  #pragma unroll
  for (int mi = 0; mi < 2; ++mi)
    #pragma unroll
    for (int ni = 0; ni < 4; ++ni) {
      const int n = ntile * TN + waveN * 64 + ni * 16 + rsel;
      #pragma unroll
      for (int r = 0; r < 8; ++r) {
        const int m = waveM * 32 + mi * 16 + mhalf + r;
        const float sc = sclS[m];
        if (sc != 0.0f)
          atomicAdd(&out[(size_t)tokS[m] * HDIM + n], sc * acc[mi][ni][r]);
      }
    }
}

// ---------------- launch ----------------
extern "C" void kernel_launch(void* const* d_in, const int* in_sizes, int n_in,
                              void* d_out, int out_size, void* d_ws, size_t ws_size,
                              hipStream_t stream) {
  const float* x    = (const float*)d_in[0];
  // d_in[1] "weights" is unused by the reference output
  const float* tw   = (const float*)d_in[2];
  const int*   te   = (const int*)d_in[3];
  const float* w1   = (const float*)d_in[4];
  const float* w2   = (const float*)d_in[5];
  const float* bias = (const float*)d_in[6];
  float* out = (float*)d_out;

  char* ws = (char*)d_ws;
  size_t off = 0;
  auto take = [&](size_t bytes) -> char* {
    char* p = ws + off;
    off += (bytes + 255) & ~(size_t)255;
    return p;
  };
  int*   cnt      = (int*)  take((size_t)NEXP * 4);
  int*   nTiles   = (int*)  take(4);
  int*   tileExp  = (int*)  take((size_t)MAX_TILES * 4);
  int*   tileBase = (int*)  take((size_t)MAX_TILES * 4);
  int*   rowTok   = (int*)  take((size_t)NEXP * T_TOK * 4);
  float* rowScale = (float*)take((size_t)NEXP * T_TOK * 4);
  unsigned short* xb   = (unsigned short*)take((size_t)T_TOK * HDIM * 2);
  unsigned short* w1b  = (unsigned short*)take((size_t)NEXP * FDIM * HDIM * 2);
  unsigned short* w2tb = (unsigned short*)take((size_t)NEXP * FDIM * HDIM * 2);
  unsigned short* hbuf = (unsigned short*)take((size_t)MAX_TILES * TM * FDIM * 2);

  k_zero<<<1, 32, 0, stream>>>(cnt, nTiles);
  k_route<<<(T_TOK + 255) / 256, 256, 0, stream>>>(te, tw, cnt, rowTok, rowScale);
  k_tiles<<<1, 1, 0, stream>>>(cnt, tileExp, tileBase, nTiles);
  k_prep_x_out<<<(T_TOK * HDIM + 255) / 256, 256, 0, stream>>>(x, bias, xb, out);
  k_prep_w<<<(int)(((size_t)NEXP * FDIM * HDIM + 255) / 256), 256, 0, stream>>>(w1, w2, w1b, w2tb);

  dim3 g1(MAX_TILES, FDIM / TN);   // data-dependent tiles handled via early-exit on nTiles
  k_gemm1<<<g1, 256, 0, stream>>>(xb, w1b, hbuf, rowTok, cnt, tileExp, tileBase, nTiles);
  dim3 g2(MAX_TILES, HDIM / TN);
  k_gemm2<<<g2, 256, 0, stream>>>(hbuf, w2tb, out, rowTok, rowScale, cnt, tileExp, tileBase, nTiles);
}
